// Decoder_154618823142
// MI455X (gfx1250) — matrix operand order repaired and checked by reading, not compiled, not run on hardware
//
#include <hip/hip_runtime.h>

typedef unsigned short u16;
typedef __attribute__((ext_vector_type(16))) __bf16 v16bf;
typedef __attribute__((ext_vector_type(8)))  float  v8f;
typedef int v4i __attribute__((vector_size(16)));   // matches async-LDS builtin param type

#define B_    32
#define TENC  256
#define TMEL  512
#define ENC_  512
#define PRE_  256
#define RNN_  1024
#define ATT_  1024
#define DEC_  1024
#define NMEL_ 80
#define NF_   32
#define KCONV 31

#define MEL_OUT_SZ  (B_ * TMEL * NMEL_)        /* 1310720 */
#define GATE_OFF    MEL_OUT_SZ
#define GATE_SZ     (TMEL * B_)                /* 16384 */
#define ALIGN_OFF   (GATE_OFF + GATE_SZ)       /* 1327104 */

#define AS1 __attribute__((address_space(1)))
#define AS3 __attribute__((address_space(3)))

#if defined(__AMDGCN__) && defined(__has_builtin)
#if __has_builtin(__builtin_amdgcn_global_load_async_to_lds_b128) && \
    __has_builtin(__builtin_amdgcn_s_wait_asynccnt)
#define HAVE_ASYNC_LDS 1
#endif
#endif
#ifndef HAVE_ASYNC_LDS
#define HAVE_ASYNC_LDS 0
#endif

__device__ __forceinline__ u16 f2b(float f) {
  unsigned u = __float_as_uint(f);
  unsigned r = ((u >> 16) & 1u) + 0x7FFFu;   // round-to-nearest-even
  return (u16)((u + r) >> 16);
}

// ---------------------------------------------------------------------------
// Generic bf16 WMMA GEMM: C[M,N] = A[M,K](bf16, row-major, lda) @ W[N,K]^T
// FLAGS: bit0 = add bias[N], bit1 = ReLU, bit2 = store bf16 (else f32)
// Block = 128 threads (4 waves); each wave owns one 16x16 N-tile, block owns
// one 16-row M-tile. A k-chunks (16x32 bf16 = 1KB) staged in LDS via async
// copies (double buffered) and shared by the 4 waves.
// Requires: M%16==0, K%32==0; N tiles clamped (inactive waves skip store).
// ---------------------------------------------------------------------------
template <int FLAGS>
__global__ __launch_bounds__(128) void gemm_bf16_wmma(
    const u16* __restrict__ A, int lda,
    const u16* __restrict__ W,
    const float* __restrict__ bias,
    void* __restrict__ Cout,
    int M, int N, int K) {
  __shared__ __attribute__((aligned(16))) u16 sA[2][512];   // 2 x 1KB
  const int lane = threadIdx.x & 31;
  const int wv   = threadIdx.x >> 5;
  const int mbase = blockIdx.y << 4;
  int nbase = (blockIdx.x * 4 + wv) << 4;
  const bool nactive = (nbase < N);
  if (!nactive) nbase = N - 16;          // clamp: compute garbage, skip store
  const int r16 = lane & 15;
  const int kh  = lane >> 4;
  const int nk  = K >> 5;

  v8f acc;
#pragma unroll
  for (int i = 0; i < 8; ++i) acc[i] = 0.0f;

  // B fragment base: lane r16 = column n, kh selects K 0-7 & 16-23 / 8-15 & 24-31
  const u16* wrow = W + (size_t)(nbase + r16) * K + (kh << 3);
#if !HAVE_ASYNC_LDS
  // Fallback: A fragments straight from global (L2-resident activations)
  const u16* arow = A + (size_t)(mbase + r16) * lda + kh * 8;
#endif

  union Frag { v16bf v; uint4 q[2]; };

#if HAVE_ASYNC_LDS
  auto stage = [&](int kc, int buf) {
    if (threadIdx.x < 32) {
#pragma unroll
      for (int i = 0; i < 2; ++i) {
        int lin = threadIdx.x * 16 + i * 512;      // byte offset in 1KB chunk
        int row = lin >> 6;                        // 64 B per A row
        int off = lin & 63;
        const u16* g = A + (size_t)(mbase + row) * lda + kc * 32 + (off >> 1);
        __builtin_amdgcn_global_load_async_to_lds_b128(
            (AS1 v4i*)(void*)g,
            (AS3 v4i*)(void*)((char*)&sA[buf][0] + lin), 0, 0);
      }
    }
  };
  stage(0, 0);
  if (threadIdx.x < 32) __builtin_amdgcn_s_wait_asynccnt(0);
  __syncthreads();
#endif

  for (int kc = 0; kc < nk; ++kc) {
    Frag a, b;
#if HAVE_ASYNC_LDS
    const int buf = kc & 1;
    if (kc + 1 < nk) stage(kc + 1, buf ^ 1);
    const u16* s = &sA[buf][0];
    // 16-bit A fragment layout: lane half 0 -> K 0-7 & 16-23, half 1 -> 8-15 & 24-31
    a.q[0] = *(const uint4*)(s + r16 * 32 + kh * 8);
    a.q[1] = *(const uint4*)(s + r16 * 32 + 16 + kh * 8);
#else
    a.q[0] = *(const uint4*)(arow + kc * 32);
    a.q[1] = *(const uint4*)(arow + kc * 32 + 16);
#endif
    const u16* wp = wrow + kc * 32;
    b.q[0] = *(const uint4*)(wp);
    b.q[1] = *(const uint4*)(wp + 16);
    acc = __builtin_amdgcn_wmma_f32_16x16x32_bf16(
        false, a.v, false, b.v, (short)0, acc, false, false);
#if HAVE_ASYNC_LDS
    if (kc + 1 < nk) {
      if (threadIdx.x < 32) __builtin_amdgcn_s_wait_asynccnt(0);
    }
    __syncthreads();
#endif
  }

  if (nactive) {
    float* cf = (float*)Cout;
    u16*   cb = (u16*)Cout;
    const int ncol = nbase + r16;
    const float bv = (FLAGS & 1) ? bias[ncol] : 0.0f;
#pragma unroll
    for (int r = 0; r < 8; ++r) {
      float v = acc[r] + bv;
      if (FLAGS & 2) v = fmaxf(v, 0.0f);
      size_t o = (size_t)(mbase + kh * 8 + r) * N + ncol;
      if (FLAGS & 4) cb[o] = f2b(v); else cf[o] = v;
    }
  }
}

// ---------------------------------------------------------------------------
// Small helper kernels
// ---------------------------------------------------------------------------
__global__ void k_zero_f32(float* p, int n) {
  int i = blockIdx.x * blockDim.x + threadIdx.x;
  if (i < n) p[i] = 0.0f;
}
__global__ void k_zero_b16(u16* p, int n) {
  int i = blockIdx.x * blockDim.x + threadIdx.x;
  if (i < n) p[i] = 0;
}
__global__ void k_f32_to_b16(u16* dst, const float* src, int n) {
  int i = blockIdx.x * blockDim.x + threadIdx.x;
  if (i < n) dst[i] = f2b(src[i]);
}
// scatter-convert src[rows,cols] f32 into dst at (rowoff, coloff), row pitch dld
__global__ void k_cat_b16(u16* dst, int dld, int rowoff, int coloff,
                          const float* src, int rows, int cols) {
  int i = blockIdx.x * blockDim.x + threadIdx.x;
  if (i < rows * cols) {
    int r = i / cols, c = i % cols;
    dst[(size_t)(r + rowoff) * dld + coloff + c] = f2b(src[i]);
  }
}
// teacher-forcing frames: mel_gt[B,80,T] -> melb[(t*B+b), 96] bf16, shifted,
// K padded 80->96 with zeros
__global__ void k_build_melb(u16* dst, const float* mel_gt, int n) {
  int i = blockIdx.x * blockDim.x + threadIdx.x;
  if (i >= n) return;
  int k = i % 96, r = i / 96;
  int b = r % B_, t = r / B_;
  float v = 0.0f;
  if (k < NMEL_ && t > 0) v = mel_gt[((size_t)b * NMEL_ + k) * TMEL + (t - 1)];
  dst[i] = f2b(v);
}
// pack [32, ldk] bf16 activation row: optional bf16 seg then up to 3 f32 segs
__global__ void k_pack_act(u16* dst, int ldk,
                           const u16* s0, int w0,
                           const float* f1, int w1,
                           const float* f2, int w2,
                           const float* f3, int w3, int total) {
  int i = blockIdx.x * blockDim.x + threadIdx.x;
  if (i >= total) return;
  int b = i / ldk, k = i % ldk;
  u16 v;
  if (k < w0) v = s0[b * w0 + k];
  else {
    int k1 = k - w0;
    if (k1 < w1) v = f2b(f1[b * w1 + k1]);
    else {
      int k2 = k1 - w1;
      if (k2 < w2) v = f2b(f2[b * w2 + k2]);
      else v = f3 ? f2b(f3[b * w3 + (k2 - w2)]) : 0;
    }
  }
  dst[i] = v;
}
// LSTM cell pointwise update from fused gates [32,4096] (i,f,g,o blocks)
__global__ void k_lstm_update(const float* __restrict__ gates,
                              float* __restrict__ h, float* __restrict__ c) {
  int i = blockIdx.x * blockDim.x + threadIdx.x;
  if (i >= B_ * RNN_) return;
  int b = i >> 10, u = i & 1023;
  const float* g = gates + (size_t)b * 4096;
  float gi = g[u], gf = g[1024 + u], gg = g[2048 + u], go = g[3072 + u];
  float si = 1.0f / (1.0f + expf(-gi));
  float sf = 1.0f / (1.0f + expf(-gf));
  float so = 1.0f / (1.0f + expf(-go));
  float cn = sf * c[i] + si * tanhf(gg);
  c[i] = cn;
  h[i] = so * tanhf(cn);
}
__global__ void k_pack_projb(float* dst, const float* proj_b, const float* gate_b) {
  int i = threadIdx.x;
  if (i < 96) dst[i] = (i < NMEL_) ? proj_b[i] : ((i == NMEL_) ? gate_b[0] : 0.0f);
}
// Location-sensitive attention energies: one wave per (b, te)
__global__ __launch_bounds__(256) void k_energy(
    const float* __restrict__ q,     // [32,1024]
    const float* __restrict__ Wenc,  // [32*256,1024]
    const float* __restrict__ aw, const float* __restrict__ awc,  // [32,256]
    const float* __restrict__ convw, // [32,2,31]
    const float* __restrict__ dense, // [1024,32]
    const float* __restrict__ vw,    // [1024]
    const int* __restrict__ tok,     // [32]
    float* __restrict__ e) {         // [32,256]
  __shared__ float convs[8][32];
  int lane = threadIdx.x & 31, wv = threadIdx.x >> 5;
  int pair = blockIdx.x * 8 + wv;
  int b = pair >> 8, t = pair & 255;
  // conv: filter f = lane over 2 channels x 31 taps, 'same' padding
  float cv = 0.0f;
#pragma unroll
  for (int ch = 0; ch < 2; ++ch) {
    const float* in = ch ? (awc + b * 256) : (aw + b * 256);
    const float* wf = convw + lane * 62 + ch * 31;
    for (int k = 0; k < KCONV; ++k) {
      int tt = t + k - (KCONV / 2);
      if (tt >= 0 && tt < TENC) cv += in[tt] * wf[k];
    }
  }
  convs[wv][lane] = cv;
  __syncthreads();
  const float* qb = q + b * ATT_;
  const float* wb = Wenc + (size_t)(b * 256 + t) * ATT_;
  float s = 0.0f;
  for (int j = 0; j < 32; ++j) {
    int a = j * 32 + lane;
    const float* dr = dense + a * NF_;
    float loc = 0.0f;
#pragma unroll
    for (int f = 0; f < NF_; ++f) loc += dr[f] * convs[wv][f];
    s += vw[a] * tanhf(qb[a] + wb[a] + loc);
  }
  for (int off = 16; off; off >>= 1) s += __shfl_xor(s, off);
  if (lane == 0) e[b * 256 + t] = (t < tok[b]) ? s : -__builtin_inff();
}
// softmax over Te + cumulative weights + context + alignment output
__global__ __launch_bounds__(256) void k_softmax_ctx(
    const float* __restrict__ e, const float* __restrict__ enc,  // [32,256,512]
    float* __restrict__ aw, float* __restrict__ awc,
    float* __restrict__ ctx, float* __restrict__ out_align, int tstep) {
  __shared__ float sh[256];
  __shared__ float sa[256];
  int b = blockIdx.x, t = threadIdx.x;
  float ev = e[b * 256 + t];
  sh[t] = ev; __syncthreads();
  for (int s = 128; s; s >>= 1) { if (t < s) sh[t] = fmaxf(sh[t], sh[t + s]); __syncthreads(); }
  float mx = sh[0]; __syncthreads();
  float ex = expf(ev - mx);
  sh[t] = ex; __syncthreads();
  for (int s = 128; s; s >>= 1) { if (t < s) sh[t] += sh[t + s]; __syncthreads(); }
  float w = ex / sh[0];
  sa[t] = w;
  aw[b * 256 + t] = w;
  awc[b * 256 + t] += w;
  out_align[(size_t)b * TMEL * TENC + (size_t)tstep * TENC + t] = w;
  __syncthreads();
  for (int d = t; d < ENC_; d += 256) {
    float acc = 0.0f;
    const float* eb = enc + (size_t)b * TENC * ENC_ + d;
    for (int tt = 0; tt < TENC; ++tt) acc += sa[tt] * eb[(size_t)tt * ENC_];
    ctx[b * ENC_ + d] = acc;
  }
}
// split fused [32,96] (mel|gate|pad) into the reference's output layouts
__global__ void k_scatter_out(const float* __restrict__ tmp,
                              float* __restrict__ out, int tstep) {
  int i = blockIdx.x * blockDim.x + threadIdx.x;
  if (i >= B_ * 96) return;
  int b = i / 96, j = i % 96;
  float v = tmp[i];
  if (j < NMEL_)
    out[(size_t)b * TMEL * NMEL_ + (size_t)tstep * NMEL_ + j] = v;  // mels [B,T,80,1]
  else if (j == NMEL_)
    out[GATE_OFF + tstep * B_ + b] = v;                             // gates flat [T,B]
}

// ---------------------------------------------------------------------------
extern "C" void kernel_launch(void* const* d_in, const int* in_sizes, int n_in,
                              void* d_out, int out_size, void* d_ws, size_t ws_size,
                              hipStream_t stream) {
  (void)in_sizes; (void)n_in; (void)out_size; (void)ws_size;
  const float* enc_out  = (const float*)d_in[0];
  const float* mel_gt   = (const float*)d_in[1];
  const int*   tok_len  = (const int*)d_in[2];
  const float* prenet_w0 = (const float*)d_in[3];   // [256,80]
  const float* prenet_w1 = (const float*)d_in[4];   // [256,256]
  const float* att_wi    = (const float*)d_in[5];   // [4096,768]
  const float* att_wh    = (const float*)d_in[6];   // [4096,1024]
  const float* att_b     = (const float*)d_in[7];   // [4096]
  const float* dec_wi    = (const float*)d_in[8];   // [4096,1536]
  const float* dec_wh    = (const float*)d_in[9];   // [4096,1024]
  const float* dec_b     = (const float*)d_in[10];  // [4096]
  const float* query_w   = (const float*)d_in[11];  // [1024,1024]
  const float* memory_w  = (const float*)d_in[12];  // [1024,512]
  const float* loc_conv  = (const float*)d_in[13];  // [32,2,31]
  const float* loc_dense = (const float*)d_in[14];  // [1024,32]
  const float* v_w       = (const float*)d_in[15];  // [1024]
  const float* proj_w    = (const float*)d_in[16];  // [80,1536]
  const float* proj_b    = (const float*)d_in[17];  // [80]
  const float* gate_w    = (const float*)d_in[18];  // [1,1536]
  const float* gate_b    = (const float*)d_in[19];  // [1]
  float* out = (float*)d_out;

  // ---- workspace carve-up (bf16 weights + f32 state) ----
  char* base = (char*)d_ws; size_t cur = 0;
  auto alloc = [&](size_t bytes) -> void* {
    void* p = base + cur; cur = (cur + bytes + 255) & ~(size_t)255; return p;
  };
  u16* attW  = (u16*)alloc((size_t)4096 * 1792 * 2);  // [4096, 768|1024]
  u16* decW  = (u16*)alloc((size_t)4096 * 2560 * 2);  // [4096, 1536|1024]
  u16* qW    = (u16*)alloc((size_t)1024 * 1024 * 2);
  u16* memW  = (u16*)alloc((size_t)1024 * 512 * 2);
  u16* w0b   = (u16*)alloc((size_t)256 * 96 * 2);     // K padded 80->96
  u16* w1b   = (u16*)alloc((size_t)256 * 256 * 2);
  u16* projW = (u16*)alloc((size_t)96 * 1536 * 2);    // rows: 80 mel | gate | pad
  float* projb = (float*)alloc(96 * 4);
  u16* encb  = (u16*)alloc((size_t)8192 * 512 * 2);
  float* Wenc = (float*)alloc((size_t)8192 * 1024 * 4);
  u16* melb  = (u16*)alloc((size_t)16384 * 96 * 2);
  u16* x1b   = (u16*)alloc((size_t)16384 * 256 * 2);
  u16* xall  = (u16*)alloc((size_t)16384 * 256 * 2);  // prenet out, all steps
  float* states = (float*)alloc((size_t)163840 * 4);
  float* att_h = states;            float* att_c = states + 32768;
  float* dec_h = states + 65536;    float* dec_c = states + 98304;
  float* ctx   = states + 131072;   float* aw    = states + 147456;
  float* awc   = states + 155648;
  float* q     = (float*)alloc((size_t)32 * 1024 * 4);
  float* evec  = (float*)alloc((size_t)32 * 256 * 4);
  float* gates = (float*)alloc((size_t)32 * 4096 * 4);
  u16* act1 = (u16*)alloc((size_t)32 * 1792 * 2);
  u16* actq = (u16*)alloc((size_t)32 * 1024 * 2);
  u16* act2 = (u16*)alloc((size_t)32 * 2560 * 2);
  u16* act3 = (u16*)alloc((size_t)32 * 1536 * 2);
  float* tmp96 = (float*)alloc((size_t)32 * 96 * 4);

  auto g1 = [](int n) { return dim3((unsigned)((n + 255) / 256)); };

  // ---- one-time (per call) prep: weights -> bf16, L2-resident ----
  k_cat_b16<<<g1(4096*768), 256, 0, stream>>>(attW, 1792, 0, 0,    att_wi, 4096, 768);
  k_cat_b16<<<g1(4096*1024),256, 0, stream>>>(attW, 1792, 0, 768,  att_wh, 4096, 1024);
  k_cat_b16<<<g1(4096*1536),256, 0, stream>>>(decW, 2560, 0, 0,    dec_wi, 4096, 1536);
  k_cat_b16<<<g1(4096*1024),256, 0, stream>>>(decW, 2560, 0, 1536, dec_wh, 4096, 1024);
  k_f32_to_b16<<<g1(1024*1024), 256, 0, stream>>>(qW,   query_w, 1024*1024);
  k_f32_to_b16<<<g1(1024*512),  256, 0, stream>>>(memW, memory_w, 1024*512);
  k_zero_b16<<<g1(256*96), 256, 0, stream>>>(w0b, 256*96);
  k_cat_b16<<<g1(256*80), 256, 0, stream>>>(w0b, 96, 0, 0, prenet_w0, 256, 80);
  k_f32_to_b16<<<g1(256*256), 256, 0, stream>>>(w1b, prenet_w1, 256*256);
  k_zero_b16<<<g1(96*1536), 256, 0, stream>>>(projW, 96*1536);
  k_cat_b16<<<g1(80*1536), 256, 0, stream>>>(projW, 1536, 0, 0, proj_w, 80, 1536);
  k_cat_b16<<<g1(1536), 256, 0, stream>>>(projW, 1536, 80, 0, gate_w, 1, 1536);
  k_pack_projb<<<1, 128, 0, stream>>>(projb, proj_b, gate_b);
  k_f32_to_b16<<<g1(8192*512), 256, 0, stream>>>(encb, enc_out, 8192*512);
  k_build_melb<<<g1(16384*96), 256, 0, stream>>>(melb, mel_gt, 16384*96);
  k_zero_f32<<<g1(163840), 256, 0, stream>>>(states, 163840);

  // ---- hoisted GEMMs: prenet (all 512 frames) and processed memory ----
  gemm_bf16_wmma<6><<<dim3(4, 1024),  128, 0, stream>>>(melb, 96,  w0b, nullptr, x1b, 16384, 256, 96);
  gemm_bf16_wmma<6><<<dim3(4, 1024),  128, 0, stream>>>(x1b, 256,  w1b, nullptr, xall, 16384, 256, 256);
  gemm_bf16_wmma<0><<<dim3(16, 512),  128, 0, stream>>>(encb, 512, memW, nullptr, Wenc, 8192, 1024, 512);

  // ---- sequential decode: 512 dependent steps ----
  for (int t = 0; t < TMEL; ++t) {
    // att-LSTM: act1 = [x_t | ctx | att_h], gates = act1 @ [Wi|Wh]^T + b
    k_pack_act<<<g1(32*1792), 256, 0, stream>>>(act1, 1792, xall + (size_t)t*32*256, 256,
                                                ctx, 512, att_h, 1024, nullptr, 0, 32*1792);
    gemm_bf16_wmma<1><<<dim3(64, 2), 128, 0, stream>>>(act1, 1792, attW, att_b, gates, 32, 4096, 1792);
    k_lstm_update<<<g1(32*1024), 256, 0, stream>>>(gates, att_h, att_c);
    // query = att_h(new) @ query_w^T
    k_pack_act<<<g1(32*1024), 256, 0, stream>>>(actq, 1024, nullptr, 0,
                                                att_h, 1024, nullptr, 0, nullptr, 0, 32*1024);
    gemm_bf16_wmma<0><<<dim3(16, 2), 128, 0, stream>>>(actq, 1024, qW, nullptr, q, 32, 1024, 1024);
    // location-sensitive attention
    k_energy<<<1024, 256, 0, stream>>>(q, Wenc, aw, awc, loc_conv, loc_dense, v_w, tok_len, evec);
    k_softmax_ctx<<<32, 256, 0, stream>>>(evec, enc_out, aw, awc, ctx, out + ALIGN_OFF, t);
    // dec-LSTM: act2 = [att_h | ctx | dec_h]
    k_pack_act<<<g1(32*2560), 256, 0, stream>>>(act2, 2560, nullptr, 0,
                                                att_h, 1024, ctx, 512, dec_h, 1024, 32*2560);
    gemm_bf16_wmma<1><<<dim3(64, 2), 128, 0, stream>>>(act2, 2560, decW, dec_b, gates, 32, 4096, 2560);
    k_lstm_update<<<g1(32*1024), 256, 0, stream>>>(gates, dec_h, dec_c);
    // projection + gate (fused [96,1536] GEMM)
    k_pack_act<<<g1(32*1536), 256, 0, stream>>>(act3, 1536, nullptr, 0,
                                                dec_h, 1024, ctx, 512, nullptr, 0, 32*1536);
    gemm_bf16_wmma<1><<<dim3(2, 2), 128, 0, stream>>>(act3, 1536, projW, projb, tmp96, 32, 96, 1536);
    k_scatter_out<<<g1(32*96), 256, 0, stream>>>(tmp96, out, t);
  }
}
